// CRFScaleModel_20091857011536
// MI455X (gfx1250) — compile-verified, matrix operand order checked by reading, not run on hardware
//
#include <hip/hip_runtime.h>
#include <stdint.h>
#include <math.h>

// Problem constants (from reference)
#define Bsz  16
#define Tlen 128
#define Kst  1024
#define Dft  768
#define KPv  102
#define NEGBIG (-1e30f)

typedef __attribute__((ext_vector_type(16))) __bf16       v16bf_t;
typedef __attribute__((ext_vector_type(8)))  float        v8f_t;
typedef __attribute__((ext_vector_type(4)))  unsigned int v4u_t;
typedef __attribute__((ext_vector_type(8)))  int          v8i_t;
typedef __attribute__((ext_vector_type(4)))  int          v4i_t;

// ---------------- workspace layout (bytes) ----------------
#define WS_EM      (0)                       // 2048*1024*4   = 8,388,608  emission f32
#define WS_TRANS   (8388608)                 // 1024*1024*4   = 4,194,304  transition f32
#define WS_EXPT    (12582912)                // 1024*1024*2   = 2,097,152  swizzled exp(trans-colmax) bf16
#define WS_XB      (14680064)                // 2048*768*2    = 3,145,728  x bf16
#define WS_SB      (17825792)                // 1024*768*2    = 1,572,864  state bf16
#define WS_CMAX    (19398656)                // 1024*4
#define WS_ALAST   (19402752)                // 16*1024*4
#define WS_LOGZ    (19468288)                // 16*4
#define WS_LOGZA   (19468352)                // 16*4
#define WS_TOPV    (19468416)                // 2048*102*4 = 835,584
#define WS_TOPI    (20304000)                // 835,584
#define WS_BP      (21139584)                // 127*16*1024*4 = 8,323,072
#define WS_CRF     (29462656)                // 2048*4 int
#define WS_ROWLSE  (29470848)                // 2048*4
#define WS_ROWARG  (29479040)                // 2048*4 int
#define WS_ROWMAXP (29487232)                // 2048*4
#define WS_ACC     (29495424)                // 64*4 scalars

// ---------------- WMMA helpers (ISA 7.12.2 layouts) ----------------
__device__ __forceinline__ int wmma_kidx(int v, int p, int hi) {
  // 16-bit A/B fragment: VGPR v in 0..7, packed half p in 0..1, hi = lane>=16
  int vv   = (v < 4) ? v : (v - 4);
  int base = (v < 4) ? 0 : 16;
  return base + hi * 8 + 2 * vv + p;
}

__device__ __forceinline__ v16bf_t frag_rowmajor(const __bf16* base, int row, int ld, int hi) {
  v16bf_t f;
#pragma unroll
  for (int v = 0; v < 8; ++v)
#pragma unroll
    for (int p = 0; p < 2; ++p) {
      int k = wmma_kidx(v, p, hi);
      f[2 * v + p] = base[(size_t)row * ld + k];
    }
  return f;
}

// fragment stored contiguously per lane (16 bf16 = 32B -> 2x b128 loads)
__device__ __forceinline__ v16bf_t frag_contig(const __bf16* p) {
  v16bf_t f;
#pragma unroll
  for (int e = 0; e < 16; ++e) f[e] = p[e];
  return f;
}

__device__ __forceinline__ v8f_t wmma_bf16(v16bf_t a, v16bf_t b, v8f_t c) {
  return __builtin_amdgcn_wmma_f32_16x16x32_bf16(false, a, false, b, (short)0, c, false, false);
}

// ---------------- TDM: tensor_load_to_lds descriptor build ----------------
__device__ __forceinline__ void tdm_load_tile_2d(unsigned lds_addr, const void* gptr,
                                                 unsigned tile_w, unsigned tile_h,
                                                 unsigned long long row_stride_elems) {
  unsigned long long ga = (unsigned long long)(uintptr_t)gptr;
  v4u_t g0;
  g0[0] = 1u;                                             // count=1, user mode
  g0[1] = lds_addr;                                       // lds_addr [63:32]
  g0[2] = (unsigned)(ga & 0xffffffffull);                 // global_addr[31:0]
  g0[3] = (unsigned)((ga >> 32) & 0x1ffffffull) | (2u << 30); // addr[56:32] | type=2
  v8i_t g1;
  g1[0] = (int)(1u << 16);                                // data_size=1 -> 2 bytes
  g1[1] = (int)((tile_w & 0xffffu) << 16);                // tensor_dim0 lo16 @bit48
  g1[2] = (int)(((tile_w >> 16) & 0xffffu) | ((tile_h & 0xffffu) << 16)); // t0 hi | t1 lo
  g1[3] = (int)(((tile_h >> 16) & 0xffffu) | ((tile_w & 0xffffu) << 16)); // t1 hi | tile_dim0
  g1[4] = (int)(tile_h & 0xffffu);                        // tile_dim1 (tile_dim2=0)
  g1[5] = (int)(row_stride_elems & 0xffffffffull);        // tensor_dim0_stride lo32
  g1[6] = (int)((row_stride_elems >> 32) & 0xffffull);    // stride hi16
  g1[7] = 0;
  v4i_t z4 = {0, 0, 0, 0};
#if __clang_major__ >= 23
  v8i_t z8 = {0, 0, 0, 0, 0, 0, 0, 0};
  __builtin_amdgcn_tensor_load_to_lds(g0, g1, z4, z4, z8, 0);
#else
  __builtin_amdgcn_tensor_load_to_lds(g0, g1, z4, z4, 0);
#endif
}

// ---------------- kernels ----------------
__global__ void cvt_bf16_kernel(const float* __restrict__ src, __bf16* __restrict__ dst, int n) {
  int i = blockIdx.x * blockDim.x + threadIdx.x;
  if (i < n) dst[i] = (__bf16)src[i];
}

// C(MxN) = A(MxKd) * B(NxKd)^T * scale, bf16 WMMA, double-buffered TDM LDS staging
__global__ void __launch_bounds__(128)
wmma_gemm_abt_kernel(const __bf16* __restrict__ A, const __bf16* __restrict__ Bm,
                     float* __restrict__ C, int M, int N, int Kd, float scale) {
  __shared__ __bf16 tA[2][64 * 32];
  __shared__ __bf16 tB[2][64 * 32];
  const int tid = threadIdx.x, wid = tid >> 5, lane = tid & 31;
  const int hi = (lane >> 4) & 1, l15 = lane & 15;
  const int row0 = blockIdx.y * 64, col0 = blockIdx.x * 64;
  v8f_t acc[4];
#pragma unroll
  for (int nt = 0; nt < 4; ++nt)
#pragma unroll
    for (int e = 0; e < 8; ++e) acc[nt][e] = 0.f;

  const int nk = Kd / 32;
  // preload k-chunk 0 into buffer 0
  if (wid == 0) {
    tdm_load_tile_2d((unsigned)(uintptr_t)(&tA[0][0]), A + (size_t)row0 * Kd, 32u, 64u,
                     (unsigned long long)Kd);
    tdm_load_tile_2d((unsigned)(uintptr_t)(&tB[0][0]), Bm + (size_t)col0 * Kd, 32u, 64u,
                     (unsigned long long)Kd);
    __builtin_amdgcn_s_wait_tensorcnt(0);
  }
  __syncthreads();

  for (int ki = 0; ki < nk; ++ki) {
    const int cur = ki & 1;
    // prefetch next chunk into the other buffer while computing this one
    if (wid == 0 && ki + 1 < nk) {
      int k1 = (ki + 1) * 32;
      tdm_load_tile_2d((unsigned)(uintptr_t)(&tA[1 - cur][0]), A + (size_t)row0 * Kd + k1,
                       32u, 64u, (unsigned long long)Kd);
      tdm_load_tile_2d((unsigned)(uintptr_t)(&tB[1 - cur][0]), Bm + (size_t)col0 * Kd + k1,
                       32u, 64u, (unsigned long long)Kd);
    }
    v16bf_t af = frag_rowmajor(&tA[cur][0], wid * 16 + l15, 32, hi);
#pragma unroll
    for (int nt = 0; nt < 4; ++nt) {
      v16bf_t bf_ = frag_rowmajor(&tB[cur][0], nt * 16 + l15, 32, hi);
      acc[nt] = wmma_bf16(af, bf_, acc[nt]);
    }
    __syncthreads();                       // all LDS reads of `cur` complete
    if (wid == 0 && ki + 1 < nk) __builtin_amdgcn_s_wait_tensorcnt(0);
    __syncthreads();                       // next buffer ready for everyone
  }
  // C/D layout: lane 0-15 -> rows r, lanes 16-31 -> rows r+8; N = lane&15
#pragma unroll
  for (int nt = 0; nt < 4; ++nt)
#pragma unroll
    for (int r = 0; r < 8; ++r) {
      int m = row0 + wid * 16 + (hi ? r + 8 : r);
      int n = col0 + nt * 16 + l15;
      C[(size_t)m * N + n] = acc[nt][r] * scale;
    }
}

__global__ void colmax_kernel(const float* __restrict__ trans, float* __restrict__ cmax) {
  int j = blockIdx.x * blockDim.x + threadIdx.x;
  if (j >= Kst) return;
  float m = NEGBIG;
  for (int i = 0; i < Kst; ++i) m = fmaxf(m, trans[(size_t)i * Kst + j]);
  cmax[j] = m;
}

// Build exp(trans - colmax) in WMMA-B-fragment-ready swizzled layout:
// expTswz[((nt*32 + kb)*32 + lane)*16 + e]  holds B[k][n] with
// k = kb*32 + kidx(e>>1, e&1, lane>>4), n = nt*16 + (lane&15).
__global__ void expt_swizzle_kernel(const float* __restrict__ trans,
                                    const float* __restrict__ cmax,
                                    __bf16* __restrict__ expTswz) {
  int idx = blockIdx.x * blockDim.x + threadIdx.x;
  if (idx >= Kst * Kst) return;
  int e    = idx & 15;
  int lane = (idx >> 4) & 31;
  int kb   = (idx >> 9) & 31;
  int nt   = idx >> 14;
  int hi = lane >> 4, l15 = lane & 15;
  int k = kb * 32 + wmma_kidx(e >> 1, e & 1, hi);
  int n = nt * 16 + l15;
  expTswz[idx] = (__bf16)__expf(trans[(size_t)k * Kst + n] - cmax[n]);
}

// Exact forward: single WG, 32 waves. Per step: 16x1024x1024 bf16 WMMA GEMM in exp space.
__global__ void __launch_bounds__(1024)
forward_exact_kernel(const float* __restrict__ em, const __bf16* __restrict__ expTswz,
                     const float* __restrict__ cmax, const int* __restrict__ y_lens,
                     float* __restrict__ aLast, float* __restrict__ logZ) {
  extern __shared__ char smraw[];
  float*  alpha = (float*)smraw;                 // 16*1024 f32
  __bf16* eA    = (__bf16*)(smraw + 65536);      // 16*1024 bf16
  float*  red   = (float*)(smraw + 98304);       // 1024 f32
  const int tid = threadIdx.x, wid = tid >> 5, lane = tid & 31;
  const int hi = (lane >> 4) & 1, l15 = lane & 15;
  const int bg = tid >> 6, sg = tid & 63;

  // t = 0 : alpha = emission[:,0,:]
  for (int r = 0; r < Bsz; ++r)
    alpha[r * Kst + tid] = em[(size_t)r * Tlen * Kst + tid];
  __syncthreads();
  for (int r = 0; r < Bsz; ++r)
    if (y_lens[r] == 1) aLast[r * Kst + tid] = alpha[r * Kst + tid];

  for (int t = 1; t < Tlen; ++t) {
    __syncthreads();
    // per-batch row max m_b (64 threads per batch)
    float lm = NEGBIG;
#pragma unroll
    for (int q = 0; q < 16; ++q) lm = fmaxf(lm, alpha[bg * Kst + sg + 64 * q]);
    red[tid] = lm;
    __syncthreads();
    for (int s = 32; s >= 1; s >>= 1) {
      if (sg < s) red[tid] = fmaxf(red[tid], red[tid + s]);
      __syncthreads();
    }
    // eA = bf16(exp(alpha - m_b))
    for (int r = 0; r < Bsz; ++r) {
      float mb = red[r * 64];
      eA[r * Kst + tid] = (__bf16)__expf(alpha[r * Kst + tid] - mb);
    }
    __syncthreads();
    // P = eA(16x1024) @ expT(1024x1024); wave handles 2 N-tiles
#pragma unroll
    for (int u = 0; u < 2; ++u) {
      int ntile = wid * 2 + u;
      int ncol  = ntile * 16 + l15;
      v8f_t acc;
#pragma unroll
      for (int e = 0; e < 8; ++e) acc[e] = 0.f;
      for (int kbi = 0; kbi < Kst / 32; ++kbi) {
        v16bf_t af  = frag_rowmajor(eA + kbi * 32, l15, Kst, hi);
        v16bf_t bf_ = frag_contig(expTswz + (((size_t)ntile * 32 + kbi) * 32 + lane) * 16);
        acc = wmma_bf16(af, bf_, acc);
      }
      float cj = cmax[ncol];
#pragma unroll
      for (int r = 0; r < 8; ++r) {
        int b = hi ? r + 8 : r;
        float nv = red[b * 64] + cj + __logf(fmaxf(acc[r], 1e-30f)) +
                   em[(size_t)b * Tlen * Kst + (size_t)t * Kst + ncol];
        alpha[b * Kst + ncol] = nv;
        if (t == y_lens[b] - 1) aLast[b * Kst + ncol] = nv;
      }
    }
  }
  // logZ[b] = logsumexp(aLast[b])
  __syncthreads();
  float lm = NEGBIG;
#pragma unroll
  for (int q = 0; q < 16; ++q) lm = fmaxf(lm, aLast[bg * Kst + sg + 64 * q]);
  red[tid] = lm;
  __syncthreads();
  for (int s = 32; s >= 1; s >>= 1) {
    if (sg < s) red[tid] = fmaxf(red[tid], red[tid + s]);
    __syncthreads();
  }
  float M = red[bg * 64];
  __syncthreads();
  float ls = 0.f;
#pragma unroll
  for (int q = 0; q < 16; ++q) ls += __expf(aLast[bg * Kst + sg + 64 * q] - M);
  red[tid] = ls;
  __syncthreads();
  for (int s = 32; s >= 1; s >>= 1) {
    if (sg < s) red[tid] += red[tid + s];
    __syncthreads();
  }
  if (sg == 0) logZ[bg] = M + __logf(red[tid]);
}

// top-KP selection per emission row (iterative argmax)
__global__ void __launch_bounds__(256)
topk_kernel(const float* __restrict__ em, float* __restrict__ topv, int* __restrict__ topi) {
  __shared__ float vals[Kst];
  __shared__ float rv[256];
  __shared__ int   ri[256];
  const int row = blockIdx.x, tid = threadIdx.x;
  for (int r = 0; r < 4; ++r) vals[tid + 256 * r] = em[(size_t)row * Kst + tid + 256 * r];
  __syncthreads();
  for (int it = 0; it < KPv; ++it) {
    float bm = NEGBIG; int bi = 0;
    for (int r = 0; r < 4; ++r) {
      int jj = tid + 256 * r;
      float v = vals[jj];
      if (v > bm) { bm = v; bi = jj; }
    }
    rv[tid] = bm; ri[tid] = bi;
    __syncthreads();
    for (int s = 128; s >= 1; s >>= 1) {
      if (tid < s) {
        if (rv[tid + s] > rv[tid] || (rv[tid + s] == rv[tid] && ri[tid + s] < ri[tid])) {
          rv[tid] = rv[tid + s]; ri[tid] = ri[tid + s];
        }
      }
      __syncthreads();
    }
    if (tid == 0) {
      topv[(size_t)row * KPv + it] = rv[0];
      topi[(size_t)row * KPv + it] = ri[0];
      vals[ri[0]] = NEGBIG;
    }
    __syncthreads();
  }
}

// Approx forward: einsum over selected state rows == gather from precomputed transition
__global__ void __launch_bounds__(128)
approx_forward_kernel(const float* __restrict__ topv, const int* __restrict__ topi,
                      const float* __restrict__ trans, const int* __restrict__ y_lens,
                      float* __restrict__ logZa) {
  __shared__ float al[128];
  __shared__ int   pidx[128];
  __shared__ int   cidx[128];
  __shared__ float aL[128];
  __shared__ float red[128];
  const int b = blockIdx.x, j = threadIdx.x, len = y_lens[b];
  if (j < KPv) {
    al[j]   = topv[((size_t)b * Tlen) * KPv + j];
    pidx[j] = topi[((size_t)b * Tlen) * KPv + j];
  } else al[j] = NEGBIG;
  aL[j] = NEGBIG;
  __syncthreads();
  if (len == 1) aL[j] = al[j];
  for (int t = 1; t < Tlen; ++t) {
    if (j < KPv) cidx[j] = topi[((size_t)b * Tlen + t) * KPv + j];
    __syncthreads();
    float nv = NEGBIG;
    if (j < KPv) {
      int cj = cidx[j];
      float Mx = NEGBIG, S = 0.f;
      for (int i = 0; i < KPv; ++i) {
        float v = al[i] + trans[(size_t)pidx[i] * Kst + cj];
        if (v > Mx) { S = S * __expf(Mx - v) + 1.f; Mx = v; }
        else        { S += __expf(v - Mx); }
      }
      nv = Mx + __logf(fmaxf(S, 1e-30f)) + topv[((size_t)b * Tlen + t) * KPv + j];
    }
    __syncthreads();
    al[j] = (j < KPv) ? nv : NEGBIG;
    if (j < KPv) pidx[j] = cidx[j];
    __syncthreads();
    if (t == len - 1) aL[j] = al[j];
  }
  // logsumexp over aL[0..KP)
  red[j] = (j < KPv) ? aL[j] : NEGBIG;
  __syncthreads();
  for (int s = 64; s >= 1; s >>= 1) {
    if (j < s) red[j] = fmaxf(red[j], red[j + s]);
    __syncthreads();
  }
  float M = red[0];
  __syncthreads();
  red[j] = (j < KPv) ? __expf(aL[j] - M) : 0.f;
  __syncthreads();
  for (int s = 64; s >= 1; s >>= 1) {
    if (j < s) red[j] += red[j + s];
    __syncthreads();
  }
  if (j == 0) logZa[b] = M + __logf(red[0]);
}

__device__ int block_argmax1024(float* rv, int* ri, int j, float v) {
  rv[j] = v; ri[j] = j;
  __syncthreads();
  for (int s = 512; s >= 1; s >>= 1) {
    if (j < s) {
      if (rv[j + s] > rv[j] || (rv[j + s] == rv[j] && ri[j + s] < ri[j])) {
        rv[j] = rv[j + s]; ri[j] = ri[j + s];
      }
    }
    __syncthreads();
  }
  int r = ri[0];
  __syncthreads();
  return r;
}

__global__ void __launch_bounds__(1024)
viterbi_kernel(const float* __restrict__ em, const float* __restrict__ trans,
               const int* __restrict__ y_lens, int* __restrict__ bp, int* __restrict__ crf) {
  __shared__ float al[Kst];
  __shared__ float rv[Kst];
  __shared__ int   ri[Kst];
  const int b = blockIdx.x, j = threadIdx.x, len = y_lens[b];
  int lastSt = 0;
  al[j] = em[(size_t)b * Tlen * Kst + j];
  __syncthreads();
  if (len == 1) lastSt = block_argmax1024(rv, ri, j, al[j]);
  for (int t = 1; t < Tlen; ++t) {
    float best = NEGBIG; int bi = 0;
    for (int i = 0; i < Kst; ++i) {
      if (i + 8 < Kst) __builtin_prefetch(&trans[(size_t)(i + 8) * Kst + j], 0, 0);
      float v = al[i] + trans[(size_t)i * Kst + j];
      if (v > best) { best = v; bi = i; }
    }
    float nv = best + em[(size_t)b * Tlen * Kst + (size_t)t * Kst + j];
    bp[((size_t)(t - 1) * Bsz + b) * Kst + j] = bi;
    __syncthreads();
    al[j] = nv;
    __syncthreads();
    if (t == len - 1) lastSt = block_argmax1024(rv, ri, j, al[j]);
  }
  __syncthreads();
  if (j == 0) {
    int s = 0;
    for (int t = Tlen - 1; t >= 0; --t) {
      int sn;
      if (t == len - 1)      sn = lastSt;
      else if (t < len - 1)  sn = bp[((size_t)t * Bsz + b) * Kst + s];
      else                   sn = 0;
      s = sn;
      crf[b * Tlen + t] = s;
    }
  }
}

__global__ void __launch_bounds__(256)
rowstats_kernel(const float* __restrict__ em, float* __restrict__ rowlse,
                int* __restrict__ rowarg, float* __restrict__ rowmaxp) {
  __shared__ float rv[256];
  __shared__ int   ri[256];
  const int row = blockIdx.x, tid = threadIdx.x;
  float bm = NEGBIG; int bi = 0;
  for (int r = 0; r < 4; ++r) {
    int jj = tid + 256 * r;
    float v = em[(size_t)row * Kst + jj];
    if (v > bm) { bm = v; bi = jj; }
  }
  rv[tid] = bm; ri[tid] = bi;
  __syncthreads();
  for (int s = 128; s >= 1; s >>= 1) {
    if (tid < s) {
      if (rv[tid + s] > rv[tid] || (rv[tid + s] == rv[tid] && ri[tid + s] < ri[tid])) {
        rv[tid] = rv[tid + s]; ri[tid] = ri[tid + s];
      }
    }
    __syncthreads();
  }
  float M = rv[0]; int arg = ri[0];
  __syncthreads();
  float s0 = 0.f;
  for (int r = 0; r < 4; ++r) s0 += __expf(em[(size_t)row * Kst + tid + 256 * r] - M);
  rv[tid] = s0;
  __syncthreads();
  for (int s = 128; s >= 1; s >>= 1) {
    if (tid < s) rv[tid] += rv[tid + s];
    __syncthreads();
  }
  if (tid == 0) {
    float lse = M + __logf(rv[0]);
    rowlse[row] = lse;
    rowarg[row] = arg;
    rowmaxp[row] = __expf(M - lse);
  }
}

__global__ void zero_acc_kernel(float* __restrict__ acc) {
  if (threadIdx.x < 64) acc[threadIdx.x] = 0.f;
}

__global__ void accum_kernel(const float* __restrict__ em, const float* __restrict__ trans,
                             const int* __restrict__ y, const int* __restrict__ y_lens,
                             const int* __restrict__ crf, const float* __restrict__ rowlse,
                             const int* __restrict__ rowarg, const float* __restrict__ rowmaxp,
                             float* __restrict__ acc) {
  int idx = blockIdx.x * blockDim.x + threadIdx.x;
  if (idx >= Bsz * Tlen) return;
  int b = idx / Tlen, t = idx % Tlen, len = y_lens[b];
  int yv = y[idx];
  if (t < len) {
    float emy = em[(size_t)b * Tlen * Kst + (size_t)t * Kst + yv];
    atomicAdd(&acc[b], emy);
    atomicAdd(&acc[32], emy - rowlse[idx]);
    atomicAdd(&acc[33], rowmaxp[idx]);
    int cp = crf[idx], sp = rowarg[idx];
    if (yv > 0)            { atomicAdd(&acc[34], 1.f); atomicAdd(&acc[38], 1.f); }
    if (cp > 0)              atomicAdd(&acc[35], 1.f);
    if (cp > 0 && yv > 0)    atomicAdd(&acc[36], 1.f);
    if (cp == yv)            atomicAdd(&acc[37], 1.f);
    if (sp > 0)              atomicAdd(&acc[39], 1.f);
    if (sp > 0 && yv > 0)    atomicAdd(&acc[40], 1.f);
    if (sp == yv)            atomicAdd(&acc[41], 1.f);
  }
  if (t < Tlen - 1 && t < len - 1)
    atomicAdd(&acc[16 + b], trans[(size_t)yv * Kst + y[idx + 1]]);
}

__global__ void final_kernel(const float* __restrict__ acc, const float* __restrict__ logZ,
                             const float* __restrict__ logZa, const int* __restrict__ y_lens,
                             const int* __restrict__ crf, float* __restrict__ out) {
  int tid = threadIdx.x;
  for (int i = tid; i < Bsz * Tlen; i += blockDim.x) out[1 + i] = (float)crf[i];
  if (tid == 0) {
    float slz = 0.f, slza = 0.f, sex = 0.f, sap = 0.f, total = 0.f;
    for (int b = 0; b < Bsz; ++b) {
      float lyp = acc[b] + acc[16 + b];
      slz += logZ[b]; slza += logZa[b];
      sex += lyp - logZ[b]; sap += lyp - logZa[b];
      total += (float)y_lens[b];
    }
    float seq_ex = -sex / (float)Bsz, seq_ap = -sap / (float)Bsz;
    float seq_local = acc[32] / total;
    float maxp = acc[33] / total;
    float loss = seq_ap + 0.1f * seq_local;
    float csup = acc[34], cpred = acc[35], cov = acc[36], ccor = acc[37];
    float cacc = ccor / total;
    float cprec = (cpred > 0.f) ? cov / fmaxf(cpred, 1.f) : 0.f;
    float crecl = cov / fmaxf(csup, 1.f);
    float cf1 = (cprec > 0.f) ? 2.f * cprec * crecl / fmaxf(cprec + crecl, 1e-12f) : 0.f;
    float ssup = acc[38], spred = acc[39], sov = acc[40], scor = acc[41];
    float sacc = scor / total;
    float sprec = (spred > 0.f) ? sov / fmaxf(spred, 1.f) : 0.f;
    float srecl = sov / fmaxf(ssup, 1.f);
    float sf1 = (sprec > 0.f) ? 2.f * sprec * srecl / fmaxf(sprec + srecl, 1e-12f) : 0.f;
    out[0]    = loss;
    out[2049] = slz / (float)Bsz;
    out[2050] = slza / (float)Bsz;
    out[2051] = seq_ex;
    out[2052] = seq_local;
    out[2053] = maxp;
    out[2054] = cacc;
    out[2055] = cf1;
    out[2056] = sacc;
    out[2057] = sf1;
  }
}

// ---------------- host launch ----------------
extern "C" void kernel_launch(void* const* d_in, const int* in_sizes, int n_in,
                              void* d_out, int out_size, void* d_ws, size_t ws_size,
                              hipStream_t stream) {
  (void)in_sizes; (void)n_in; (void)out_size; (void)ws_size;
  const float* x_emb = (const float*)d_in[0];
  const float* state = (const float*)d_in[1];
  const int*   y     = (const int*)d_in[2];
  const int*   ylens = (const int*)d_in[3];
  float* out = (float*)d_out;
  char* ws = (char*)d_ws;

  float*  em      = (float*)(ws + WS_EM);
  float*  trans   = (float*)(ws + WS_TRANS);
  __bf16* expT    = (__bf16*)(ws + WS_EXPT);
  __bf16* xb      = (__bf16*)(ws + WS_XB);
  __bf16* sb      = (__bf16*)(ws + WS_SB);
  float*  cmax    = (float*)(ws + WS_CMAX);
  float*  aLast   = (float*)(ws + WS_ALAST);
  float*  logZ    = (float*)(ws + WS_LOGZ);
  float*  logZa   = (float*)(ws + WS_LOGZA);
  float*  topv    = (float*)(ws + WS_TOPV);
  int*    topi    = (int*)(ws + WS_TOPI);
  int*    bp      = (int*)(ws + WS_BP);
  int*    crf     = (int*)(ws + WS_CRF);
  float*  rowlse  = (float*)(ws + WS_ROWLSE);
  int*    rowarg  = (int*)(ws + WS_ROWARG);
  float*  rowmaxp = (float*)(ws + WS_ROWMAXP);
  float*  acc     = (float*)(ws + WS_ACC);

  const float scale = 1.0f / sqrtf((float)Dft);
  const int nX = Bsz * Tlen * Dft, nS = Kst * Dft;

  cvt_bf16_kernel<<<(nX + 255) / 256, 256, 0, stream>>>(x_emb, xb, nX);
  cvt_bf16_kernel<<<(nS + 255) / 256, 256, 0, stream>>>(state, sb, nS);

  dim3 ge(Kst / 64, (Bsz * Tlen) / 64);
  wmma_gemm_abt_kernel<<<ge, 128, 0, stream>>>(xb, sb, em, Bsz * Tlen, Kst, Dft, scale);
  dim3 gt(Kst / 64, Kst / 64);
  wmma_gemm_abt_kernel<<<gt, 128, 0, stream>>>(sb, sb, trans, Kst, Kst, Dft, scale);

  colmax_kernel<<<Kst / 256, 256, 0, stream>>>(trans, cmax);
  expt_swizzle_kernel<<<(Kst * Kst) / 256, 256, 0, stream>>>(trans, cmax, expT);

  forward_exact_kernel<<<1, 1024, 102400, stream>>>(em, expT, cmax, ylens, aLast, logZ);

  topk_kernel<<<Bsz * Tlen, 256, 0, stream>>>(em, topv, topi);
  approx_forward_kernel<<<Bsz, 128, 0, stream>>>(topv, topi, trans, ylens, logZa);

  viterbi_kernel<<<Bsz, 1024, 0, stream>>>(em, trans, ylens, bp, crf);

  rowstats_kernel<<<Bsz * Tlen, 256, 0, stream>>>(em, rowlse, rowarg, rowmaxp);

  zero_acc_kernel<<<1, 64, 0, stream>>>(acc);
  accum_kernel<<<(Bsz * Tlen + 255) / 256, 256, 0, stream>>>(em, trans, y, ylens, crf,
                                                             rowlse, rowarg, rowmaxp, acc);
  final_kernel<<<1, 256, 0, stream>>>(acc, logZ, logZa, ylens, crf, out);
}
